// Decoder_64269890617388
// MI455X (gfx1250) — matrix operand order repaired and checked by reading, not compiled, not run on hardware
//
#include <hip/hip_runtime.h>
#include <hip/hip_bf16.h>

typedef __bf16 bf16;
typedef __attribute__((ext_vector_type(8)))  bf16  bfx8;
typedef __attribute__((ext_vector_type(16))) bf16  v16bf;
typedef __attribute__((ext_vector_type(8)))  float v8f;

#define BATCH  16384
#define TSTEPS 12
#define KDIM   1056   // MLP + ZD
#define NCAT   512    // 128 (h0) + 384 (3H gates)
#define HDIM   128
#define G3     384

__device__ __forceinline__ v16bf cat8(bfx8 lo, bfx8 hi) {
  return __builtin_shufflevector(lo, hi, 0,1,2,3,4,5,6,7,8,9,10,11,12,13,14,15);
}
__device__ __forceinline__ v8f splat8(float x) {
  return (v8f){x,x,x,x,x,x,x,x};
}

// ---------------- Phase 0a: pack [Wdec ; Wih[:, :1056]] -> bf16, fold biases ----------------
__global__ __launch_bounds__(256) void k_pack(const float* __restrict__ Wdec,
                                              const float* __restrict__ bdec,
                                              const float* __restrict__ Wih,
                                              const float* __restrict__ bih,
                                              bf16* __restrict__ wcat,
                                              float* __restrict__ biascat) {
  int idx = blockIdx.x * 256 + threadIdx.x;
  if (idx < NCAT * KDIM) {
    int n = idx / KDIM, k = idx - n * KDIM;
    float v = (n < HDIM) ? Wdec[(size_t)n * KDIM + k]
                         : Wih[(size_t)(n - HDIM) * 1060 + k];
    wcat[idx] = (bf16)v;
  }
  if (idx < NCAT) biascat[idx] = (idx < HDIM) ? bdec[idx] : bih[idx - HDIM];
}

// ---------------- Phase 0b: a_seq [T,B,2] and gated sg features [T,B,2] ----------------
__global__ __launch_bounds__(256) void k_small(const float* __restrict__ lst,
                                               const float* __restrict__ lpos,
                                               const float* __restrict__ sg,
                                               const float* __restrict__ fut,
                                               const int*   __restrict__ upd,
                                               const float* __restrict__ Wto,
                                               const float* __restrict__ bto,
                                               const float* __restrict__ Wsg,
                                               const float* __restrict__ bsg,
                                               float* __restrict__ aseq,
                                               float* __restrict__ sgf) {
  int b = blockIdx.x * 256 + threadIdx.x;
  if (b >= BATCH) return;
  // a0 = last_obs_st @ W_to_vel.T + b_to_vel
  float ax = bto[0], ay = bto[1];
  #pragma unroll
  for (int j = 0; j < 6; ++j) { float s = lst[b*6 + j]; ax += s*Wto[j]; ay += s*Wto[6 + j]; }
  aseq[b*2 + 0] = ax; aseq[b*2 + 1] = ay;
  for (int t = 1; t < TSTEPS; ++t) {
    aseq[(size_t)t*BATCH*2 + b*2 + 0] = fut[(size_t)(t-1)*BATCH*6 + b*6 + 2];
    aseq[(size_t)t*BATCH*2 + b*2 + 1] = fut[(size_t)(t-1)*BATCH*6 + b*6 + 3];
  }
  const float invDT = 1.0f / 1.6f;  // DT = 0.4*4
  float px = lpos[b*2 + 0], py = lpos[b*2 + 1];
  float qx[5], qy[5];
  qx[0] = 0.f; qy[0] = 0.f;
  #pragma unroll
  for (int j = 1; j <= 4; ++j) {
    qx[j] = (sg[b*8 + (j-1)*2 + 0] - px) * 0.01f;
    qy[j] = (sg[b*8 + (j-1)*2 + 1] - py) * 0.01f;
  }
  float f0[4], f1[4];
  #pragma unroll
  for (int j = 0; j < 4; ++j) {
    float vx = (qx[j+1] - qx[j]) * invDT;
    float vy = (qy[j+1] - qy[j]) * invDT;
    f0[j] = bsg[0] + qx[j+1]*Wsg[0] + qy[j+1]*Wsg[1] + vx*Wsg[2] + vy*Wsg[3];
    f1[j] = bsg[1] + qx[j+1]*Wsg[4] + qy[j+1]*Wsg[5] + vx*Wsg[6] + vy*Wsg[7];
  }
  int u1 = upd[1], u2 = upd[2], u3 = upd[3];
  for (int t = 0; t < TSTEPS; ++t) {
    int j = 0;
    if (u1 <= t) j = 1;
    if (u2 <= t) j = 2;
    if (u3 <= t) j = 3;
    sgf[(size_t)t*BATCH*2 + b*2 + 0] = f0[j];
    sgf[(size_t)t*BATCH*2 + b*2 + 1] = f1[j];
  }
}

// ---------------- Phase 1: C[B,512] = zx[B,1056] @ Wcat.T + bias  (bf16 WMMA) ----------------
// Block: 256 threads (8 waves), 128 batch rows.  zx panel staged bf16 in 264KB LDS.
__global__ __launch_bounds__(256) void k_gemm(const float* __restrict__ enc,
                                              const float* __restrict__ zin,
                                              const bf16*  __restrict__ wcat,
                                              const float* __restrict__ biascat,
                                              float* __restrict__ Cout) {
  extern __shared__ char smem[];
  bf16* sZX = (bf16*)smem;                  // [128][1064] (padded stride)
  const int tid  = threadIdx.x;
  const int row0 = blockIdx.x * 128;
  for (int i = tid; i < 128 * KDIM; i += 256) {
    int r = i / KDIM, c = i - r * KDIM;
    float v = (c < 1024) ? enc[(size_t)(row0 + r) * 1024 + c]
                         : zin[(size_t)(row0 + r) * 32 + (c - 1024)];
    sZX[r * 1064 + c] = (bf16)v;
  }
  __syncthreads();
  const int wave = tid >> 5, lane = tid & 31;
  const int lm = lane & 15, lh = lane >> 4;
  for (int q = 0; q < 4; ++q) {
    const int nt = wave * 4 + q;
    const int n  = nt * 16 + lm;
    const float bias = biascat[n];
    const bf16* wrow = wcat + (size_t)n * KDIM;
    for (int mt = 0; mt < 8; ++mt) {
      v8f acc = splat8(bias);
      const int m = mt * 16 + lm;
      for (int k0 = 0; k0 < KDIM; k0 += 32) {
        const int lo = k0 + lh * 16;                      // A: lanes0-15 K0..15, lanes16-31 K16..31
        bfx8 alo = *(const bfx8*)&sZX[m * 1064 + lo];
        bfx8 ahi = *(const bfx8*)&sZX[m * 1064 + lo + 8];
        v16bf aF = cat8(alo, ahi);
        const int kb = k0 + lh * 16;                      // B: lanes0-15 K0..15, lanes16-31 K16..31 of column n
        bfx8 blo = *(const bfx8*)&wrow[kb];
        bfx8 bhi = *(const bfx8*)&wrow[kb + 8];
        v16bf bF = cat8(blo, bhi);
        acc = __builtin_amdgcn_wmma_f32_16x16x32_bf16(false, aF, false, bF, (short)0, acc, false, false);
      }
      #pragma unroll
      for (int i = 0; i < 8; ++i) {                       // D: vgpr i -> M = i + 8*lh, N = lm
        const int mm = mt * 16 + i + lh * 8;
        Cout[(size_t)(row0 + mm) * NCAT + n] = acc[i];
      }
    }
  }
}

// ---------------- Phase 2: 12-step GRU + mu/std heads (bf16 WMMA, barrier-free time loop) ----
// Block: 256 threads (8 waves), 128 batch rows; wave w owns rows [w*16, w*16+16).
__global__ __launch_bounds__(256) void k_gru(const float* __restrict__ Cbuf,
                                             const float* __restrict__ Whh,
                                             const float* __restrict__ bhh,
                                             const float* __restrict__ Wih,
                                             const float* __restrict__ aseq,
                                             const float* __restrict__ sgf,
                                             const float* __restrict__ Wmu,
                                             const float* __restrict__ bmu,
                                             const float* __restrict__ Wstd,
                                             const float* __restrict__ bstd,
                                             float* __restrict__ outp) {
  extern __shared__ char smem[];
  bf16*  sWhh = (bf16*)smem;                    // [384][136] bf16   104448 B
  bf16*  sH   = (bf16*)(smem + 104448);         // [128][136] bf16    34816 B
  float* sA   = (float*)(smem + 139264);        // [12][128][2]       12288 B
  float* sSG  = sA  + TSTEPS * 128 * 2;         // [12][128][2]       12288 B
  float* sWs  = sSG + TSTEPS * 128 * 2;         // [384][4]            6144 B
  float* sbh  = sWs + G3 * 4;                   // [384]               1536 B
  float* sWm  = sbh + G3;                       // [2][128]            1024 B
  float* sWsd = sWm + 256;                      // [2][128]            1024 B
  float* sbm  = sWsd + 256;                     // [2]
  float* sbs  = sbm + 2;                        // [2]

  const int tid  = threadIdx.x;
  const int row0 = blockIdx.x * 128;

  for (int i = tid; i < G3 * HDIM; i += 256) {
    int g = i >> 7, k = i & 127;
    sWhh[g * 136 + k] = (bf16)Whh[i];
  }
  for (int i = tid; i < HDIM * HDIM; i += 256) {
    int r = i >> 7, k = i & 127;
    sH[r * 136 + k] = (bf16)Cbuf[(size_t)(row0 + r) * NCAT + k];   // h0
  }
  for (int i = tid; i < TSTEPS * 128 * 2; i += 256) {
    int t = i >> 8, rem = i & 255;
    sA[i]  = aseq[(size_t)t * BATCH * 2 + (size_t)row0 * 2 + rem];
    sSG[i] = sgf [(size_t)t * BATCH * 2 + (size_t)row0 * 2 + rem];
  }
  for (int i = tid; i < G3 * 4; i += 256) {
    int g = i >> 2, c = i & 3;
    sWs[i] = Wih[(size_t)g * 1060 + 1056 + c];                     // small-input gate columns
  }
  for (int i = tid; i < G3; i += 256) sbh[i] = bhh[i];
  if (tid < 256) { sWm[tid] = Wmu[tid]; sWsd[tid] = Wstd[tid]; }
  if (tid < 2)   { sbm[tid] = bmu[tid]; sbs[tid]  = bstd[tid]; }
  __syncthreads();

  const int wave = tid >> 5, lane = tid & 31;
  const int lm = lane & 15, lh = lane >> 4;
  const int mt = wave;                           // this wave's 16-row tile
  float* out_mu  = outp;
  float* out_std = outp + (size_t)TSTEPS * BATCH * 2;

  for (int t = 0; t < TSTEPS; ++t) {
    v16bf aF[4];                                 // full K=128 of this wave's rows
    {
      const int m = mt * 16 + lm;
      #pragma unroll
      for (int k4 = 0; k4 < 4; ++k4) {
        const int lo = k4 * 32 + lh * 16;
        bfx8 alo = *(const bfx8*)&sH[m * 136 + lo];
        bfx8 ahi = *(const bfx8*)&sH[m * 136 + lo + 8];
        aF[k4] = cat8(alo, ahi);
      }
    }
    #pragma unroll 1
    for (int ht = 0; ht < 8; ++ht) {
      const int hcol = ht * 16 + lm;
      v8f aR = splat8(sbh[hcol]);
      v8f aZ = splat8(sbh[128 + hcol]);
      v8f aN = splat8(sbh[256 + hcol]);
      #pragma unroll
      for (int k4 = 0; k4 < 4; ++k4) {
        const int kb = k4 * 32 + lh * 16;
        bfx8 blo = *(const bfx8*)&sWhh[hcol * 136 + kb];
        bfx8 bhi = *(const bfx8*)&sWhh[hcol * 136 + kb + 8];
        aR = __builtin_amdgcn_wmma_f32_16x16x32_bf16(false, aF[k4], false, cat8(blo, bhi), (short)0, aR, false, false);
        blo = *(const bfx8*)&sWhh[(128 + hcol) * 136 + kb];
        bhi = *(const bfx8*)&sWhh[(128 + hcol) * 136 + kb + 8];
        aZ = __builtin_amdgcn_wmma_f32_16x16x32_bf16(false, aF[k4], false, cat8(blo, bhi), (short)0, aZ, false, false);
        blo = *(const bfx8*)&sWhh[(256 + hcol) * 136 + kb];
        bhi = *(const bfx8*)&sWhh[(256 + hcol) * 136 + kb + 8];
        aN = __builtin_amdgcn_wmma_f32_16x16x32_bf16(false, aF[k4], false, cat8(blo, bhi), (short)0, aN, false, false);
      }
      const float wr0 = sWs[hcol*4+0], wr1 = sWs[hcol*4+1], wr2 = sWs[hcol*4+2], wr3 = sWs[hcol*4+3];
      const float wz0 = sWs[(128+hcol)*4+0], wz1 = sWs[(128+hcol)*4+1], wz2 = sWs[(128+hcol)*4+2], wz3 = sWs[(128+hcol)*4+3];
      const float wn0 = sWs[(256+hcol)*4+0], wn1 = sWs[(256+hcol)*4+1], wn2 = sWs[(256+hcol)*4+2], wn3 = sWs[(256+hcol)*4+3];
      #pragma unroll
      for (int i = 0; i < 8; ++i) {
        const int ml = mt * 16 + i + lh * 8;
        const float* cp = Cbuf + (size_t)(row0 + ml) * NCAT;
        const float a0 = sA[t*256 + ml*2 + 0], a1 = sA[t*256 + ml*2 + 1];
        const float s0 = sSG[t*256 + ml*2 + 0], s1 = sSG[t*256 + ml*2 + 1];
        float gr  = cp[128 + hcol] + a0*wr0 + a1*wr1 + s0*wr2 + s1*wr3 + aR[i];
        float gz  = cp[256 + hcol] + a0*wz0 + a1*wz1 + s0*wz2 + s1*wz3 + aZ[i];
        float gnx = cp[384 + hcol] + a0*wn0 + a1*wn1 + s0*wn2 + s1*wn3;
        float r = 1.f / (1.f + __expf(-gr));
        float z = 1.f / (1.f + __expf(-gz));
        float n = tanhf(gnx + r * aN[i]);        // b_hh(n) lives inside aN, scaled by r (matches ref)
        float hold = (float)sH[ml * 136 + hcol];
        float hnew = (1.f - z) * n + z * hold;
        sH[ml * 136 + hcol] = (bf16)hnew;        // wave-private rows: no barrier needed
      }
    }
    // mu / std heads for this wave's 16 rows: lanes 0-15 -> mu, 16-31 -> std
    {
      const int ml = mt * 16 + lm;
      const float* W = lh ? sWsd : sWm;
      float acc0 = lh ? sbs[0] : sbm[0];
      float acc1 = lh ? sbs[1] : sbm[1];
      #pragma unroll 4
      for (int k = 0; k < 128; ++k) {
        float hv = (float)sH[ml * 136 + k];
        acc0 += hv * W[k];
        acc1 += hv * W[128 + k];
      }
      float o0 = acc0, o1 = acc1;
      if (lh) { o0 = __expf(0.5f * acc0); o1 = __expf(0.5f * acc1); }   // sqrt(exp(x))
      float* dst = lh ? out_std : out_mu;
      dst[(size_t)t * BATCH * 2 + (size_t)(row0 + ml) * 2 + 0] = o0;
      dst[(size_t)t * BATCH * 2 + (size_t)(row0 + ml) * 2 + 1] = o1;
    }
  }
}

extern "C" void kernel_launch(void* const* d_in, const int* in_sizes, int n_in,
                              void* d_out, int out_size, void* d_ws, size_t ws_size,
                              hipStream_t stream) {
  (void)in_sizes; (void)n_in; (void)out_size; (void)ws_size;
  const float* last_obs_st  = (const float*)d_in[0];
  const float* last_obs_pos = (const float*)d_in[1];
  const float* enc          = (const float*)d_in[2];
  const float* zin          = (const float*)d_in[3];
  const float* sg           = (const float*)d_in[4];
  const float* fut          = (const float*)d_in[5];
  const int*   upd          = (const int*)  d_in[6];
  const float* Wdec = (const float*)d_in[7];
  const float* bdec = (const float*)d_in[8];
  const float* Wto  = (const float*)d_in[9];
  const float* bto  = (const float*)d_in[10];
  const float* Wih  = (const float*)d_in[11];
  const float* bih  = (const float*)d_in[12];
  const float* Whh  = (const float*)d_in[13];
  const float* bhh  = (const float*)d_in[14];
  const float* Wmu  = (const float*)d_in[15];
  const float* bmu  = (const float*)d_in[16];
  const float* Wstd = (const float*)d_in[17];
  const float* bstd = (const float*)d_in[18];
  const float* Wsg  = (const float*)d_in[19];
  const float* bsg  = (const float*)d_in[20];

  char* ws = (char*)d_ws;
  bf16*  wcat    = (bf16*) ws;                    // 512*1056*2   = 1,081,344 B
  float* biascat = (float*)(ws + 1081344);        // 512*4        =     2,048 B
  float* Cbuf    = (float*)(ws + 1083392);        // 16384*512*4  = 33,554,432 B
  float* aseq    = (float*)(ws + 34637824);       // 12*16384*2*4 =  1,572,864 B
  float* sgfb    = (float*)(ws + 36210688);       // 12*16384*2*4 =  1,572,864 B  (total ~36 MB)

  // Large dynamic-LDS opt-in (CDNA5: 320 KB per workgroup)
  hipFuncSetAttribute((const void*)k_gemm, hipFuncAttributeMaxDynamicSharedMemorySize, 272384);
  hipFuncSetAttribute((const void*)k_gru,  hipFuncAttributeMaxDynamicSharedMemorySize, 173600);

  k_pack <<<(NCAT * KDIM + 255) / 256, 256, 0, stream>>>(Wdec, bdec, Wih, bih, wcat, biascat);
  k_small<<<BATCH / 256, 256, 0, stream>>>(last_obs_st, last_obs_pos, sg, fut, upd,
                                           Wto, bto, Wsg, bsg, aseq, sgfb);
  k_gemm <<<BATCH / 128, 256, 272384, stream>>>(enc, zin, wcat, biascat, Cbuf);
  k_gru  <<<BATCH / 128, 256, 173600, stream>>>(Cbuf, Whh, bhh, Wih, aseq, sgfb,
                                                Wmu, bmu, Wstd, bstd, (float*)d_out);
}
